// GGNN_7773890805947
// MI455X (gfx1250) — matrix operand order repaired and checked by reading, not compiled, not run on hardware
//
#include <hip/hip_runtime.h>
#include <hip/hip_bf16.h>

#define BATCH 32
#define NODES 1024
#define DIM   128
#define DIM2  256
#define TSTEP 5

typedef __attribute__((ext_vector_type(16))) __bf16 v16bf;
typedef __attribute__((ext_vector_type(8)))  float  v8f;

union FragU { v16bf v; uint4 q[2]; };

// ---------------------------------------------------------------------------
// CDNA5 async global->LDS copy (ASYNCcnt-tracked), builtin if present,
// inline-asm fallback emitting the same GLOBAL_LOAD_ASYNC_TO_LDS_B128.
// The builtin's params are AS(1)/AS(3) pointers to int __vector(4).
// ---------------------------------------------------------------------------
typedef int v4i_vs __attribute__((vector_size(16)));
typedef __attribute__((address_space(1))) v4i_vs* g_v4i_ptr;
typedef __attribute__((address_space(3))) v4i_vs* l_v4i_ptr;

#if __has_builtin(__builtin_amdgcn_global_load_async_to_lds_b128)
#define ASYNC_B128(g, l)                                                      \
  __builtin_amdgcn_global_load_async_to_lds_b128((g_v4i_ptr)(g),              \
                                                 (l_v4i_ptr)(l), 0, 0)
#else
#define ASYNC_B128(g, l)                                                      \
  asm volatile("global_load_async_to_lds_b128 %0, %1, off" ::                 \
               "v"((unsigned)(uintptr_t)(l)), "v"((const void*)(g)) : "memory")
#endif

#if __has_builtin(__builtin_amdgcn_s_wait_asynccnt)
#define ASYNC_WAIT() __builtin_amdgcn_s_wait_asynccnt(0)
#else
#define ASYNC_WAIT() asm volatile("s_wait_asynccnt 0" ::: "memory")
#endif

// A-fragment (16x32 bf16) from a row-major matrix: row = lane&15,
// lanes 0-15 hold K {0..7,16..23}, lanes 16-31 hold K {8..15,24..31}.
// Two contiguous 16B loads per lane (ISA 7.12.2 16-bit A layout).
__device__ __forceinline__ v16bf load_a_frag(const __bf16* base, int ld) {
  const int lane = threadIdx.x & 31;
  const int r    = lane & 15;
  const int half = lane >> 4;
  const __bf16* p = base + (size_t)r * ld + half * 8;
  FragU f;
  f.q[0] = *(const uint4*)(p);
  f.q[1] = *(const uint4*)(p + 16);
  return f.v;
}

// B-fragment (32x16 bf16) from a K-contiguous (column-major) matrix:
// element (k,c) at base[c*ld + k]. lanes 0-15: col=lane, K {0..7,16..23};
// lanes 16-31: col=lane-16, K {8..15,24..31}. Two contiguous 16B loads per lane.
__device__ __forceinline__ v16bf load_b_frag(const __bf16* base, int ld) {
  const int lane = threadIdx.x & 31;
  const int c    = lane & 15;
  const int half = lane >> 4;
  const __bf16* p = base + (size_t)c * ld + half * 8;
  FragU f;
  f.q[0] = *(const uint4*)(p);
  f.q[1] = *(const uint4*)(p + 16);
  return f.v;
}

__device__ __forceinline__ v8f wmma_bf16f32(v16bf a, v16bf b, v8f c) {
  // (neg_a, A, neg_b, B, c_mod, C, reuse_a, reuse_b)
  return __builtin_amdgcn_wmma_f32_16x16x32_bf16(false, a, false, b, (short)0, c,
                                                 false, false);
}

__device__ __forceinline__ v8f vzero() {
  v8f z = {0.f, 0.f, 0.f, 0.f, 0.f, 0.f, 0.f, 0.f};
  return z;
}

__device__ __forceinline__ float sigmoidf(float x) {
  return 1.0f / (1.0f + __expf(-x));
}

// ---------------------------------------------------------------------------
// Message-passing: for batch b, m_in = A_in * h_b, m_out = A_out * h_b.
// Block = 128 rows x 128 cols x (two adjacency matrices); 8 waves.
// The shared h-chunk (128 d x 64 m, K-contiguous) is double-buffered in LDS
// via async global->LDS copies; all waves consume B-fragments with
// ds_load_b128 while the next chunk streams in (ASYNCcnt + barrier sync).
// ---------------------------------------------------------------------------
__global__ __launch_bounds__(256) void k_msg(
    const __bf16* __restrict__ Ain, const __bf16* __restrict__ Aout,
    const __bf16* __restrict__ htr, __bf16* __restrict__ av) {
  __shared__ __bf16 tile[2][DIM * 64];   // 2 x 16KB double buffer
  const int b    = blockIdx.y;
  const int wave = threadIdx.x >> 5;
  const int n0   = blockIdx.x * 128 + wave * 16;
  const int lane = threadIdx.x & 31;
  const int c    = lane & 15;
  const int half = lane >> 4;
  const int tid  = threadIdx.x;

  const __bf16* htrb = htr + (size_t)b * DIM * NODES;
  // staging map: 1024 b128 transfers per chunk; thread covers 4 (d rows +32)
  const int sd = tid >> 3;          // d row 0..31 (+32 per j)
  const int sm = (tid & 7) * 8;     // element offset within 64-wide chunk

  v8f accI[8], accO[8];
#pragma unroll
  for (int dt = 0; dt < 8; ++dt) { accI[dt] = vzero(); accO[dt] = vzero(); }

  // preload chunk 0
#pragma unroll
  for (int j = 0; j < 4; ++j) {
    const int d = sd + 32 * j;
    ASYNC_B128(htrb + (size_t)d * NODES + sm, &tile[0][d * 64 + sm]);
  }
  ASYNC_WAIT();
  __syncthreads();

  for (int chunk = 0; chunk < NODES / 64; ++chunk) {
    const int m0  = chunk * 64;
    const int cur = chunk & 1;
    if (chunk + 1 < NODES / 64) {    // async-prefetch next chunk
#pragma unroll
      for (int j = 0; j < 4; ++j) {
        const int d = sd + 32 * j;
        ASYNC_B128(htrb + (size_t)d * NODES + m0 + 64 + sm,
                   &tile[cur ^ 1][d * 64 + sm]);
      }
      __builtin_prefetch(Ain + (size_t)n0 * NODES + m0 + 64, 0, 1);
    }
#pragma unroll
    for (int kk = 0; kk < 64; kk += 32) {
      v16bf aI = load_a_frag(Ain  + (size_t)n0 * NODES + m0 + kk, NODES);
      v16bf aO = load_a_frag(Aout + (size_t)n0 * NODES + m0 + kk, NODES);
#pragma unroll
      for (int dt = 0; dt < 8; ++dt) {
        v16bf bb = load_b_frag(&tile[cur][dt * 16 * 64 + kk], 64);
        accI[dt] = wmma_bf16f32(aI, bb, accI[dt]);
        accO[dt] = wmma_bf16f32(aO, bb, accO[dt]);
      }
    }
    ASYNC_WAIT();       // own async issues complete
    __syncthreads();    // all waves' writes visible; safe to overwrite cur
  }

  const size_t rowg = (size_t)b * NODES + n0;
#pragma unroll
  for (int dt = 0; dt < 8; ++dt) {
#pragma unroll
    for (int i = 0; i < 8; ++i) {
      const size_t row = rowg + i + 8 * half;   // C-layout: VGPR i, half: M or M+8
      av[row * DIM2 + dt * 16 + c]       = (__bf16)accI[dt][i];
      av[row * DIM2 + DIM + dt * 16 + c] = (__bf16)accO[dt][i];
    }
  }
}

// ---------------------------------------------------------------------------
// Fused GRU-style update for a 128-row strip of [B*N]:
//   u3 = h@u3_wT + u3_b ; z = sig(av@w3T + w3_b + u3) ; r = sig(av@w4T + w4_b + u3)
//   hv = tanh(av@w5T + w5_b + (r*h)@u5_wT + u5_b) ; h = (1-z)h + z*hv
// Weight B-operands use the native row-major W[d,k] layout (K-contiguous).
// (r*h) is staged bf16 in a per-wave LDS tile, then consumed as an A-fragment.
// ---------------------------------------------------------------------------
__global__ __launch_bounds__(256) void k_gate(
    const __bf16* __restrict__ avbf,
    __bf16* __restrict__ hrm, __bf16* __restrict__ htr,
    float* __restrict__ hA,
    const __bf16* __restrict__ w3bf, const __bf16* __restrict__ w4bf,
    const __bf16* __restrict__ w5bf, const __bf16* __restrict__ u3bf,
    const __bf16* __restrict__ u5bf,
    const float* __restrict__ b3, const float* __restrict__ bu3,
    const float* __restrict__ b4, const float* __restrict__ b5,
    const float* __restrict__ bu5) {
  __shared__ __bf16 lds_rh[8][16 * DIM];   // 8 waves * 4KB = 32KB
  const int wave = threadIdx.x >> 5;
  const int lane = threadIdx.x & 31;
  const int c    = lane & 15;
  const int half = lane >> 4;
  const int r0   = blockIdx.x * 128 + wave * 16;   // row in [0, B*N)

  // ---- U = h @ u3_wT (K = 128) ----
  v8f U[8];
#pragma unroll
  for (int dt = 0; dt < 8; ++dt) U[dt] = vzero();
  for (int k0 = 0; k0 < DIM; k0 += 32) {
    v16bf a = load_a_frag(hrm + (size_t)r0 * DIM + k0, DIM);
#pragma unroll
    for (int dt = 0; dt < 8; ++dt) {
      v16bf bb = load_b_frag(u3bf + (size_t)dt * 16 * DIM + k0, DIM);
      U[dt] = wmma_bf16f32(a, bb, U[dt]);
    }
  }

  // ---- S = av @ w3_wT (K = 256) -> z ----
  v8f S[8];
#pragma unroll
  for (int dt = 0; dt < 8; ++dt) S[dt] = vzero();
  for (int k0 = 0; k0 < DIM2; k0 += 32) {
    v16bf a = load_a_frag(avbf + (size_t)r0 * DIM2 + k0, DIM2);
#pragma unroll
    for (int dt = 0; dt < 8; ++dt) {
      v16bf bb = load_b_frag(w3bf + (size_t)dt * 16 * DIM2 + k0, DIM2);
      S[dt] = wmma_bf16f32(a, bb, S[dt]);
    }
  }
  v8f Z[8];
#pragma unroll
  for (int dt = 0; dt < 8; ++dt) {
    const float bw = b3[dt * 16 + c] + bu3[dt * 16 + c];
#pragma unroll
    for (int i = 0; i < 8; ++i) Z[dt][i] = sigmoidf(S[dt][i] + U[dt][i] + bw);
  }

  // ---- S = av @ w4_wT -> r; stage rh = r*h into LDS (bf16) ----
#pragma unroll
  for (int dt = 0; dt < 8; ++dt) S[dt] = vzero();
  for (int k0 = 0; k0 < DIM2; k0 += 32) {
    v16bf a = load_a_frag(avbf + (size_t)r0 * DIM2 + k0, DIM2);
#pragma unroll
    for (int dt = 0; dt < 8; ++dt) {
      v16bf bb = load_b_frag(w4bf + (size_t)dt * 16 * DIM2 + k0, DIM2);
      S[dt] = wmma_bf16f32(a, bb, S[dt]);
    }
  }
#pragma unroll
  for (int dt = 0; dt < 8; ++dt) {
    const float bw = b4[dt * 16 + c] + bu3[dt * 16 + c];
#pragma unroll
    for (int i = 0; i < 8; ++i) {
      const int roff = i + 8 * half;
      const int col  = dt * 16 + c;
      const float rr = sigmoidf(S[dt][i] + U[dt][i] + bw);
      const float hv = hA[(size_t)(r0 + roff) * DIM + col];
      lds_rh[wave][roff * DIM + col] = (__bf16)(rr * hv);
    }
  }
  // DS ops are in-order per wave; drain DScnt before cross-lane re-read.
  asm volatile("s_wait_dscnt 0" ::: "memory");

  // ---- S = av @ w5_wT + rh @ u5_wT  (single accumulator chain) ----
#pragma unroll
  for (int dt = 0; dt < 8; ++dt) S[dt] = vzero();
  for (int k0 = 0; k0 < DIM2; k0 += 32) {
    v16bf a = load_a_frag(avbf + (size_t)r0 * DIM2 + k0, DIM2);
#pragma unroll
    for (int dt = 0; dt < 8; ++dt) {
      v16bf bb = load_b_frag(w5bf + (size_t)dt * 16 * DIM2 + k0, DIM2);
      S[dt] = wmma_bf16f32(a, bb, S[dt]);
    }
  }
  for (int k0 = 0; k0 < DIM; k0 += 32) {
    v16bf a = load_a_frag(&lds_rh[wave][0] + k0, DIM);
#pragma unroll
    for (int dt = 0; dt < 8; ++dt) {
      v16bf bb = load_b_frag(u5bf + (size_t)dt * 16 * DIM + k0, DIM);
      S[dt] = wmma_bf16f32(a, bb, S[dt]);
    }
  }

  // ---- hv = tanh(S + b5 + bu5); h = (1-z)h + z*hv; emit all three h layouts ----
#pragma unroll
  for (int dt = 0; dt < 8; ++dt) {
    const float bw = b5[dt * 16 + c] + bu5[dt * 16 + c];
#pragma unroll
    for (int i = 0; i < 8; ++i) {
      const int roff = i + 8 * half;
      const int col  = dt * 16 + c;
      const size_t idx = (size_t)(r0 + roff) * DIM + col;
      const float hold = hA[idx];
      const float hv   = tanhf(S[dt][i] + bw);
      const float zz   = Z[dt][i];
      const float hn   = (1.0f - zz) * hold + zz * hv;
      hA[idx]  = hn;
      hrm[idx] = (__bf16)hn;
      const int rg = r0 + roff;
      const int bb = rg >> 10;          // / NODES
      const int nn = rg & (NODES - 1);  // % NODES
      htr[((size_t)bb * DIM + col) * NODES + nn] = (__bf16)hn;
    }
  }
}

// ---------------------------------------------------------------------------
// Prep / epilogue
// ---------------------------------------------------------------------------
__global__ __launch_bounds__(256) void k_cvt(const float* __restrict__ s,
                                             __bf16* __restrict__ d, int n) {
  int i = blockIdx.x * 256 + threadIdx.x;
  if (i < n) d[i] = (__bf16)s[i];
}

__global__ __launch_bounds__(256) void k_prep_h(const float* __restrict__ in,
                                                float* __restrict__ hA,
                                                __bf16* __restrict__ hrm,
                                                __bf16* __restrict__ htr) {
  int i = blockIdx.x * 256 + threadIdx.x;   // exactly B*N*D threads
  float v = in[i];
  hA[i]  = v;
  hrm[i] = (__bf16)v;
  int d = i & (DIM - 1);
  int n = (i >> 7) & (NODES - 1);
  int b = i >> 17;
  htr[((size_t)b * DIM + d) * NODES + n] = (__bf16)v;
}

__global__ __launch_bounds__(256) void k_out(const float* __restrict__ s,
                                             float* __restrict__ d) {
  int i = blockIdx.x * 256 + threadIdx.x;
  d[i] = s[i];
}

extern "C" void kernel_launch(void* const* d_in, const int* in_sizes, int n_in,
                              void* d_out, int out_size, void* d_ws, size_t ws_size,
                              hipStream_t stream) {
  const float* input = (const float*)d_in[0];
  const float* in_m  = (const float*)d_in[1];
  const float* out_m = (const float*)d_in[2];
  const float* w3w = (const float*)d_in[3];
  const float* w3b = (const float*)d_in[4];
  const float* u3w = (const float*)d_in[5];
  const float* u3b = (const float*)d_in[6];
  const float* w4w = (const float*)d_in[7];
  const float* w4b = (const float*)d_in[8];
  const float* w5w = (const float*)d_in[9];
  const float* w5b = (const float*)d_in[10];
  const float* u5w = (const float*)d_in[11];
  const float* u5b = (const float*)d_in[12];
  (void)in_sizes; (void)n_in; (void)out_size; (void)ws_size;

  char* ws = (char*)d_ws;
  size_t off = 0;
  float*  hA    = (float*)(ws + off);  off += (size_t)BATCH * NODES * DIM * 4;   // 16 MB
  __bf16* avbf  = (__bf16*)(ws + off); off += (size_t)BATCH * NODES * DIM2 * 2;  // 16 MB
  __bf16* hrm   = (__bf16*)(ws + off); off += (size_t)BATCH * NODES * DIM * 2;   //  8 MB
  __bf16* htr   = (__bf16*)(ws + off); off += (size_t)BATCH * NODES * DIM * 2;   //  8 MB
  __bf16* Ainb  = (__bf16*)(ws + off); off += (size_t)NODES * NODES * 2;         //  2 MB
  __bf16* Aoutb = (__bf16*)(ws + off); off += (size_t)NODES * NODES * 2;         //  2 MB
  __bf16* w3bf  = (__bf16*)(ws + off); off += (size_t)DIM * DIM2 * 2;
  __bf16* w4bf  = (__bf16*)(ws + off); off += (size_t)DIM * DIM2 * 2;
  __bf16* w5bf  = (__bf16*)(ws + off); off += (size_t)DIM * DIM2 * 2;
  __bf16* u3bf  = (__bf16*)(ws + off); off += (size_t)DIM * DIM * 2;
  __bf16* u5bf  = (__bf16*)(ws + off); off += (size_t)DIM * DIM * 2;

  // One-time bf16 conversions (adjacency, weights) + h in 3 layouts.
  k_cvt<<<NODES * NODES / 256, 256, 0, stream>>>(in_m, Ainb, NODES * NODES);
  k_cvt<<<NODES * NODES / 256, 256, 0, stream>>>(out_m, Aoutb, NODES * NODES);
  k_cvt<<<DIM * DIM2 / 256, 256, 0, stream>>>(w3w, w3bf, DIM * DIM2);
  k_cvt<<<DIM * DIM2 / 256, 256, 0, stream>>>(w4w, w4bf, DIM * DIM2);
  k_cvt<<<DIM * DIM2 / 256, 256, 0, stream>>>(w5w, w5bf, DIM * DIM2);
  k_cvt<<<DIM * DIM / 256, 256, 0, stream>>>(u3w, u3bf, DIM * DIM);
  k_cvt<<<DIM * DIM / 256, 256, 0, stream>>>(u5w, u5bf, DIM * DIM);
  k_prep_h<<<BATCH * NODES * DIM / 256, 256, 0, stream>>>(input, hA, hrm, htr);

  for (int t = 0; t < TSTEP; ++t) {
    k_msg<<<dim3(NODES / 128, BATCH), 256, 0, stream>>>(Ainb, Aoutb, htr, avbf);
    k_gate<<<BATCH * NODES / 128, 256, 0, stream>>>(avbf, hrm, htr, hA,
        w3bf, w4bf, w5bf, u3bf, u5bf, w3b, u3b, w4b, w5b, u5b);
  }
  k_out<<<BATCH * NODES * DIM / 256, 256, 0, stream>>>(hA, (float*)d_out);
}
